// MultiHeadAttention_81578608820861
// MI455X (gfx1250) — matrix operand order repaired and checked by reading, not compiled, not run on hardware
//
#include <hip/hip_runtime.h>

// MHA forward for B=2, T=2048, C=2048, H=16, Dh=128 on gfx1250 (wave32, WMMA).
// Pipeline: cvt(x,Wqkv,Wout)->bf16 | [x @ Wqkv] | causal flash-attn | [attn @ Wout]->f32.
// Matmuls: v_wmma_f32_16x16x32_bf16, f32 accumulate. GEMMs are double-buffered with
// CDNA5 async global->LDS copies (ASYNCcnt) when the toolchain exposes them.

typedef __attribute__((ext_vector_type(16))) __bf16 v16bf;
typedef __attribute__((ext_vector_type(8)))  __bf16 v8bf;
typedef __attribute__((ext_vector_type(8)))  float  v8f;

#define B_  2
#define T_  2048
#define C_  2048
#define H_  16
#define DH_ 128

#if defined(__has_builtin)
#  if __has_builtin(__builtin_amdgcn_global_load_async_to_lds_b128)
#    define HAVE_ASYNC_LDS 1
#  endif
#endif
#ifndef HAVE_ASYNC_LDS
#  define HAVE_ASYNC_LDS 0
#endif

// exact parameter types per hipcc diagnostic: int vector_size(16), AS1 / AS3
typedef int vi4 __attribute__((vector_size(16)));
typedef __attribute__((address_space(1))) vi4* gvi4p;
typedef __attribute__((address_space(3))) vi4* lvi4p;

static __device__ __forceinline__ void async_copy16(const __bf16* g, __bf16* l) {
#if HAVE_ASYNC_LDS
  __builtin_amdgcn_global_load_async_to_lds_b128((gvi4p)(g), (lvi4p)(l), 0, 0);
#else
  *(v8bf*)l = *(const v8bf*)g;
#endif
}
static __device__ __forceinline__ void async_wait0() {
#if HAVE_ASYNC_LDS
  asm volatile("s_wait_asynccnt 0" ::: "memory");
#endif
}

static __device__ __forceinline__ v16bf cat16(v8bf lo, v8bf hi) {
  v16bf r;
#pragma unroll
  for (int i = 0; i < 8; ++i) { r[i] = lo[i]; r[i + 8] = hi[i]; }
  return r;
}

// ---------------------------------------------------------------------------
// Elementwise f32 -> bf16, 8 elements/thread (one-time cost; removes cvt VALU
// and halves operand traffic in the GEMM hot loops).
// ---------------------------------------------------------------------------
__global__ __launch_bounds__(256)
void cvt_bf16(const float* __restrict__ in, __bf16* __restrict__ out, int n8) {
  const int i = blockIdx.x * 256 + threadIdx.x;
  if (i >= n8) return;
  const float4* p = (const float4*)in + (size_t)i * 2;
  const float4 a = p[0], b = p[1];
  v8bf o;
  o[0] = (__bf16)a.x; o[1] = (__bf16)a.y; o[2] = (__bf16)a.z; o[3] = (__bf16)a.w;
  o[4] = (__bf16)b.x; o[5] = (__bf16)b.y; o[6] = (__bf16)b.z; o[7] = (__bf16)b.w;
  ((v8bf*)out)[i] = o;
}

// ---------------------------------------------------------------------------
// Tiled GEMM: Out[M,N] = A[M,K] @ Bw[K,N], both bf16.  BM=BN=128, BK=32,
// 256 thr / 8 waves; wave grid 2x4, each wave a 64x32 tile = 4x2 fragments.
// Double-buffered LDS: A staged via async global->LDS, B staged transposed
// ([n][k]) via VGPRs so both operand fragment reads are contiguous b128s.
// Requires M%128==0, N%128==0, K%32==0.
// ---------------------------------------------------------------------------
template <bool OUT_F32>
__global__ __launch_bounds__(256)
void gemm_wmma(const __bf16* __restrict__ A, const __bf16* __restrict__ Bw,
               void* __restrict__ Outp, int M, int N, int K) {
  constexpr int BM = 128, BN = 128, BK = 32;
  constexpr int LDA = BK + 8;   // 40 bf16 -> 80B row stride, conflict-free reads
  constexpr int LDB = BK + 8;
  __shared__ __bf16 As[2][BM * LDA];
  __shared__ __bf16 Bst[2][BN * LDB];

  const int tid  = threadIdx.x;
  const int lane = tid & 31;
  const int wid  = tid >> 5;
  const int l16  = lane & 15;
  const int hi   = lane >> 4;
  const int kb   = hi * 8;

  const int n0 = blockIdx.x * BN;
  const int m0 = blockIdx.y * BM;
  const int wm0 = (wid & 1) * 64;
  const int wn0 = (wid >> 1) * 32;

  const v8f vzero = {};
  v8f acc[4][2];
#pragma unroll
  for (int i = 0; i < 4; ++i)
#pragma unroll
    for (int j = 0; j < 2; ++j) acc[i][j] = vzero;

  const int arow = tid >> 1;        // 0..127
  const int aseg = (tid & 1) * 16;  // 0 / 16
  const int brow = tid >> 3;        // 0..31 (k)
  const int bseg = (tid & 7) * 16;  // n segment

  auto stageA = [&](int buf, int k0) {
    const __bf16* src = A + (size_t)(m0 + arow) * K + k0 + aseg;
    __bf16* dst = &As[buf][arow * LDA + aseg];
    async_copy16(src, dst);
    async_copy16(src + 8, dst + 8);
  };
  auto loadB = [&](int k0, v8bf& b0, v8bf& b1) {
    const __bf16* src = Bw + (size_t)(k0 + brow) * N + n0 + bseg;
    b0 = ((const v8bf*)src)[0];
    b1 = ((const v8bf*)src)[1];
  };
  auto storeB = [&](int buf, v8bf b0, v8bf b1) {
    __bf16* base = &Bst[buf][0];
#pragma unroll
    for (int i = 0; i < 8; ++i) base[(bseg + i) * LDB + brow] = b0[i];
#pragma unroll
    for (int i = 0; i < 8; ++i) base[(bseg + 8 + i) * LDB + brow] = b1[i];
  };

  // prologue: fill buffer 0
  {
    v8bf b0, b1;
    loadB(0, b0, b1);
    stageA(0, 0);
    storeB(0, b0, b1);
    async_wait0();
    __syncthreads();
  }

  const int nsteps = K / BK;
  for (int s = 0; s < nsteps; ++s) {
    const int buf = s & 1, nxt = buf ^ 1;
    const bool more = (s + 1) < nsteps;

    v8bf b0, b1;
    if (more) {
      loadB((s + 1) * BK, b0, b1);   // global loads in flight over compute
      stageA(nxt, (s + 1) * BK);     // async copy straight into LDS
      if (s + 2 < nsteps)            // prefetch B tile after next into L2
        __builtin_prefetch(Bw + (size_t)((s + 2) * BK + brow) * N + n0 + bseg, 0, 0);
    }

    // ---- compute on buf: 6 b128 fragment loads + 8 WMMAs per wave
    v16bf afrag[4];
#pragma unroll
    for (int mf = 0; mf < 4; ++mf) {
      const __bf16* p = &As[buf][(wm0 + mf * 16 + l16) * LDA];
      afrag[mf] = cat16(*(const v8bf*)(p + hi * 16), *(const v8bf*)(p + hi * 16 + 8));
    }
#pragma unroll
    for (int nf = 0; nf < 2; ++nf) {
      const __bf16* p = &Bst[buf][(wn0 + nf * 16 + l16) * LDB + hi * 16];
      v16bf bfrag = cat16(*(const v8bf*)p, *(const v8bf*)(p + 8));
#pragma unroll
      for (int mf = 0; mf < 4; ++mf)
        acc[mf][nf] = __builtin_amdgcn_wmma_f32_16x16x32_bf16(
            false, afrag[mf], false, bfrag, (short)0, acc[mf][nf], false, false);
    }

    if (more) storeB(nxt, b0, b1);   // loadcnt wait lands after compute
    async_wait0();
    __syncthreads();
  }

  // ---- epilogue: D layout is (M = v + 8*hi, N = l16)
#pragma unroll
  for (int mf = 0; mf < 4; ++mf)
#pragma unroll
    for (int nf = 0; nf < 2; ++nf)
#pragma unroll
      for (int v = 0; v < 8; ++v) {
        const size_t r = (size_t)(m0 + wm0 + mf * 16 + v + 8 * hi);
        const size_t c = (size_t)(n0 + wn0 + nf * 16 + l16);
        if (OUT_F32) ((float*)Outp)[r * N + c] = acc[mf][nf][v];
        else         ((__bf16*)Outp)[r * N + c] = (__bf16)acc[mf][nf][v];
      }
}

// ---------------------------------------------------------------------------
// Causal flash attention over bf16 qkv workspace [B*T, 3C] (q|k|v per row).
// Grid: (T/128, B*H). WG = 256 thr = 8 waves; wave w owns q rows [qb+16w,+16).
// Uniform k-loop over 64-key tiles (legal __syncthreads); K tile staged with
// async global->LDS overlapping the V transpose staging.
// ---------------------------------------------------------------------------
__global__ __launch_bounds__(256)
void attn_fwd(const __bf16* __restrict__ qkv, __bf16* __restrict__ out) {
  constexpr int C3 = 3 * C_;
  constexpr int LDK = DH_ + 8;   // 136
  constexpr int LDV = 64 + 8;    // 72  (Vt: [d][k])
  constexpr int LDP = 64 + 8;    // 72  (per-wave P bounce)
  __shared__ __bf16 Ks[64 * LDK];
  __shared__ __bf16 Vt[DH_ * LDV];
  __shared__ __bf16 Pb[8 * 16 * LDP];

  const int tid = threadIdx.x, lane = tid & 31, wid = tid >> 5;
  const int l16 = lane & 15, hi = lane >> 4, kb = hi * 8;
  const int b = blockIdx.y >> 4, h = blockIdx.y & 15;
  const int qb = blockIdx.x * 128;
  const int q0 = qb + wid * 16;

  const __bf16* base = qkv + (size_t)b * T_ * C3;
  const __bf16* Qp = base + h * DH_;
  const __bf16* Kp = base + C_ + h * DH_;
  const __bf16* Vp = base + 2 * C_ + h * DH_;

  // preload Q A-fragments: 4 chunks of k=32 over Dh=128
  v16bf qf[4];
#pragma unroll
  for (int kk = 0; kk < 4; ++kk) {
    const __bf16* p = Qp + (size_t)(q0 + l16) * C3 + kk * 32;
    qf[kk] = cat16(*(const v8bf*)(p + hi * 16), *(const v8bf*)(p + hi * 16 + 8));
  }

  const v8f vzero = {};
  v8f oacc[8];
#pragma unroll
  for (int f = 0; f < 8; ++f) oacc[f] = vzero;
  float mstate[8], lstate[8];
#pragma unroll
  for (int v = 0; v < 8; ++v) { mstate[v] = -3.0e38f; lstate[v] = 0.0f; }

  const int jend = (qb + 127) / 64;   // inclusive, uniform per block
  const int srow = tid >> 2;          // 0..63
  const int sseg = (tid & 3) * 32;    // 0..96
  const int vd   = tid >> 1;          // 0..127 (d for V transpose staging)
  const int vks  = (tid & 1) * 32;    // 0 / 32

  for (int j = 0; j <= jend; ++j) {
    const int k0 = j * 64;
    // ---- stage K tile [64 x 128] row-major via async copy (ASYNCcnt)
    {
      const __bf16* src = Kp + (size_t)(k0 + srow) * C3 + sseg;
      __bf16* dst = &Ks[srow * LDK + sseg];
#pragma unroll
      for (int i = 0; i < 4; ++i) async_copy16(src + i * 8, dst + i * 8);
    }
    // ---- stage V transposed: Vt[d][k]  (overlaps with async K copy)
    {
      const __bf16* src = Vp + (size_t)(k0 + vks) * C3 + vd;
      __bf16* dst = &Vt[vd * LDV + vks];
#pragma unroll
      for (int i = 0; i < 32; ++i) dst[i] = src[(size_t)i * C3];
    }
    async_wait0();
    __syncthreads();

    // ---- S = Q @ K^T  (4 col-tiles of 16, 4 k-steps of 32)
    v8f s[4];
#pragma unroll
    for (int n = 0; n < 4; ++n) s[n] = vzero;
#pragma unroll
    for (int n = 0; n < 4; ++n) {
      const __bf16* pr = &Ks[(n * 16 + l16) * LDK];
#pragma unroll
      for (int kk = 0; kk < 4; ++kk) {
        const __bf16* p = pr + kk * 32 + hi * 16;
        v16bf bf = cat16(*(const v8bf*)p, *(const v8bf*)(p + 8));
        s[n] = __builtin_amdgcn_wmma_f32_16x16x32_bf16(
            false, qf[kk], false, bf, (short)0, s[n], false, false);
      }
    }

    // ---- scale, causal mask, online softmax (row lives in 16 lanes of a half)
    const float scale = 0.08838834764831845f;  // 1/sqrt(128)
    float alpha[8];
#pragma unroll
    for (int v = 0; v < 8; ++v) {
      const int qi = q0 + v + 8 * hi;
      float mx = -3.0e38f;
#pragma unroll
      for (int n = 0; n < 4; ++n) {
        float val = s[n][v] * scale;
        const int kj = k0 + n * 16 + l16;
        val = (kj <= qi) ? val : -3.0e38f;
        s[n][v] = val;
        mx = fmaxf(mx, val);
      }
#pragma unroll
      for (int off = 1; off < 16; off <<= 1)
        mx = fmaxf(mx, __shfl_xor(mx, off, 32));
      const float mnew = fmaxf(mstate[v], mx);
      alpha[v] = __expf(mstate[v] - mnew);
      float rs = 0.0f;
#pragma unroll
      for (int n = 0; n < 4; ++n) {
        float p = __expf(s[n][v] - mnew);
        s[n][v] = p;
        rs += p;
      }
#pragma unroll
      for (int off = 1; off < 16; off <<= 1)
        rs += __shfl_xor(rs, off, 32);
      lstate[v] = lstate[v] * alpha[v] + rs;
      mstate[v] = mnew;
    }
    // rescale running O
#pragma unroll
    for (int f = 0; f < 8; ++f)
#pragma unroll
      for (int v = 0; v < 8; ++v) oacc[f][v] *= alpha[v];

    // ---- P (D-layout) -> per-wave LDS -> A-layout fragments
    __bf16* Pw = &Pb[wid * 16 * LDP];
#pragma unroll
    for (int v = 0; v < 8; ++v)
#pragma unroll
      for (int n = 0; n < 4; ++n)
        Pw[(v + 8 * hi) * LDP + n * 16 + l16] = (__bf16)s[n][v];
    asm volatile("s_wait_dscnt 0" ::: "memory");
    __builtin_amdgcn_wave_barrier();

    // ---- O += P @ V   (P: 16x64, V: 64x128 -> 8 d-fragments)
#pragma unroll
    for (int kk2 = 0; kk2 < 2; ++kk2) {
      const __bf16* pp = &Pw[l16 * LDP + kk2 * 32];
      v16bf pa = cat16(*(const v8bf*)(pp + hi * 16), *(const v8bf*)(pp + hi * 16 + 8));
#pragma unroll
      for (int f = 0; f < 8; ++f) {
        const __bf16* vp = &Vt[(f * 16 + l16) * LDV + kk2 * 32 + hi * 16];
        v16bf vb = cat16(*(const v8bf*)vp, *(const v8bf*)(vp + 8));
        oacc[f] = __builtin_amdgcn_wmma_f32_16x16x32_bf16(
            false, pa, false, vb, (short)0, oacc[f], false, false);
      }
    }
    __syncthreads();
  }

  // ---- normalize and store [B*T, C] bf16
  __bf16* obase = out + (size_t)b * T_ * C_ + h * DH_;
#pragma unroll
  for (int v = 0; v < 8; ++v) {
    const float inv = 1.0f / lstate[v];
    const int qi = q0 + v + 8 * hi;
#pragma unroll
    for (int f = 0; f < 8; ++f)
      obase[(size_t)qi * C_ + f * 16 + l16] = (__bf16)(oacc[f][v] * inv);
  }
}

// ---------------------------------------------------------------------------
extern "C" void kernel_launch(void* const* d_in, const int* in_sizes, int n_in,
                              void* d_out, int out_size, void* d_ws, size_t ws_size,
                              hipStream_t stream) {
  const float* x    = (const float*)d_in[0];
  // d_in[1] = causal tril mask: computed analytically in attn_fwd, unused.
  const float* Wqkv = (const float*)d_in[2];
  const float* Wout = (const float*)d_in[3];
  float* out = (float*)d_out;

  // workspace layout (bf16):
  //  qkv [B*T,3C] 50.3MB | attn [B*T,C] 16.8MB | xb [B*T,C] 16.8MB
  //  wqkvb [C,3C] 25.2MB | woutb [C,C] 8.4MB            total ~117MB
  char* ws = (char*)d_ws;
  __bf16* qkv   = (__bf16*)ws;                                   ws += (size_t)B_ * T_ * 3 * C_ * 2;
  __bf16* attn  = (__bf16*)ws;                                   ws += (size_t)B_ * T_ * C_ * 2;
  __bf16* xb    = (__bf16*)ws;                                   ws += (size_t)B_ * T_ * C_ * 2;
  __bf16* wqkvb = (__bf16*)ws;                                   ws += (size_t)C_ * 3 * C_ * 2;
  __bf16* woutb = (__bf16*)ws;

  dim3 blk(256);
  const int nx = B_ * T_ * C_, nq = C_ * 3 * C_, nw = C_ * C_;
  cvt_bf16<<<dim3(nx / 8 / 256), blk, 0, stream>>>(x, xb, nx / 8);
  cvt_bf16<<<dim3(nq / 8 / 256), blk, 0, stream>>>(Wqkv, wqkvb, nq / 8);
  cvt_bf16<<<dim3(nw / 8 / 256), blk, 0, stream>>>(Wout, woutb, nw / 8);

  gemm_wmma<false><<<dim3((3 * C_) / 128, (B_ * T_) / 128), blk, 0, stream>>>(
      xb, wqkvb, qkv, B_ * T_, 3 * C_, C_);
  attn_fwd<<<dim3(T_ / 128, B_ * H_), blk, 0, stream>>>(qkv, attn);
  gemm_wmma<true><<<dim3(C_ / 128, (B_ * T_) / 128), blk, 0, stream>>>(
      attn, woutb, out, B_ * T_, C_, C_);
}
